// SparseFormer_65850438582903
// MI455X (gfx1250) — matrix operand order repaired and checked by reading, not compiled, not run on hardware
//
#include <hip/hip_runtime.h>
#include <hip/hip_bf16.h>
#include <math.h>

// ---------------------------------------------------------------------------
// SparseFormer forward for MI455X (gfx1250, wave32, WMMA f16 16x16x32).
//   B=64 N=64 DIM=256 HEADS=4 P=16 H=W=56 C=256 CH=64 IN_PTS=OUT_PTS=64
// Workspace (f16 unless noted):
//   sampled_h [4096][64][64]            32 MB   (row-major per token)
//   hidden_h  [4096][64]                512 KB
//   x2_h      [4096][4096]              32 MB   (row-major per token)
//   W2p       packed B-frags of W2       1 MB
//   Woutp     packed B-frags of W_out    2 MB
// Packed B-fragment layout: frag f -> 32 lanes x 16 halves (32B/lane),
//   element j of lane l = src[(kt*32 + ((l>>4)<<4) + j)*N + nt*16 + (l&15)]
// ---------------------------------------------------------------------------

typedef __attribute__((ext_vector_type(16))) _Float16 v16h;
typedef __attribute__((ext_vector_type(8)))  _Float16 v8h;
typedef __attribute__((ext_vector_type(8)))  float    v8f;
typedef __attribute__((ext_vector_type(4)))  int      v4i;

__device__ __forceinline__ v8f wmma_f16(v16h a, v16h b, v8f c) {
  return __builtin_amdgcn_wmma_f32_16x16x32_f16(false, a, false, b, (short)0, c,
                                                false, false);
}

// A fragment (16x32 f16) from row-major [*, ld] matrix (global or LDS).
// lane<16 -> M=lane, K={k0..k0+15}
// lane>=16 -> M=lane-16, K={k0+16..k0+31}
__device__ __forceinline__ v16h load_a_f16(const _Float16* p, int ld, int row0,
                                           int k0, int lane, int validRows) {
  const int m  = lane & 15;
  const int kb = k0 + ((lane >> 4) << 4);
  v16h a;
  if (m < validRows) {
    const _Float16* r = p + (row0 + m) * ld + kb;
    v8h lo = *(const v8h*)(r);
    v8h hi = *(const v8h*)(r + 8);
#pragma unroll
    for (int j = 0; j < 8; ++j) { a[j] = lo[j]; a[j + 8] = hi[j]; }
  } else {
#pragma unroll
    for (int j = 0; j < 16; ++j) a[j] = (_Float16)0.f;
  }
  return a;
}

// B fragment from packed fragment-major storage: two b128 loads per lane.
__device__ __forceinline__ v16h load_b_packed(const _Float16* p, int fragIdx,
                                              int lane) {
  const _Float16* q = p + ((size_t)fragIdx * 32 + lane) * 16;
  v8h lo = *(const v8h*)q;
  v8h hi = *(const v8h*)(q + 8);
  v16h b;
#pragma unroll
  for (int j = 0; j < 8; ++j) { b[j] = lo[j]; b[j + 8] = hi[j]; }
  return b;
}

__device__ __forceinline__ float gelu_exact(float x) {
  return 0.5f * x * (1.f + erff(x * 0.70710678118654752f));
}

// Async global->LDS copy of 16 bytes (CDNA5 GLOBAL_LOAD_ASYNC_TO_LDS_B128).
// Toolchain signature (from probe diagnostic): first arg is v4i* in the
// global address space; pass both pointers as address-space-qualified v4i*.
__device__ __forceinline__ void async_copy_b128(const _Float16* gsrc,
                                                _Float16* lds_dst) {
#if __has_builtin(__builtin_amdgcn_global_load_async_to_lds_b128)
  __builtin_amdgcn_global_load_async_to_lds_b128(
      (__attribute__((address_space(1))) v4i*)(unsigned long long)gsrc,
      (__attribute__((address_space(3))) v4i*)(unsigned)(unsigned long long)lds_dst,
      0, 0);
#else
  unsigned l = (unsigned)(unsigned long long)lds_dst;
  asm volatile("global_load_async_to_lds_b128 %0, %1, off"
               :: "v"(l), "v"(gsrc) : "memory");
#endif
}
__device__ __forceinline__ void wait_async0() {
#if __has_builtin(__builtin_amdgcn_s_wait_asynccnt)
  __builtin_amdgcn_s_wait_asynccnt(0);
#else
  asm volatile("s_wait_asynccnt 0" ::: "memory");
#endif
}

// ---------------------------------------------------------------------------
// K0: pack a row-major f32 K x N weight matrix into f16 B-fragment layout.
// One thread per (frag, lane) slot; writes 32 contiguous bytes.
// ---------------------------------------------------------------------------
__global__ void __launch_bounds__(256)
k_pack_b(const float* __restrict__ src, _Float16* __restrict__ dst, int K, int N) {
  const int idx = blockIdx.x * blockDim.x + threadIdx.x;
  const int totalLanes = (K >> 5) * (N >> 4) * 32;
  if (idx >= totalLanes) return;
  const int l = idx & 31;
  const int f = idx >> 5;
  const int ntiles = N >> 4;
  const int kt = f / ntiles, nt = f % ntiles;
  const int n  = nt * 16 + (l & 15);
  const int kb = kt * 32 + ((l >> 4) << 4);
  v8h lo, hi;
#pragma unroll
  for (int j = 0; j < 8; ++j) lo[j] = (_Float16)src[(size_t)(kb + j) * N + n];
#pragma unroll
  for (int j = 0; j < 8; ++j) hi[j] = (_Float16)src[(size_t)(kb + 8 + j) * N + n];
  v8h* out = (v8h*)(dst + (size_t)idx * 16);
  out[0] = lo;
  out[1] = hi;
}

// ---------------------------------------------------------------------------
// K1: offsets -> point normalize -> bilinear sample.  1 block per token.
// ---------------------------------------------------------------------------
__global__ void __launch_bounds__(128)
k_sample(const float* __restrict__ img_feat, const float* __restrict__ token_roi,
         const float* __restrict__ query, const float* __restrict__ W_off,
         const float* __restrict__ b_off, _Float16* __restrict__ sampled_h) {
  __shared__ float sq[256];
  __shared__ float soff[128];
  __shared__ float spts[128];
  __shared__ float sstat[4];  // mean_x, 1/(3*std_x), mean_y, 1/(3*std_y)
  const int tid = threadIdx.x;
  const int tok = blockIdx.x;
  const int b   = tok >> 6;

  for (int i = tid; i < 256; i += 128) sq[i] = query[tok * 256 + i];
  __syncthreads();

  float acc = b_off[tid];  // tid = p*2 + coord
  for (int i = 0; i < 256; ++i) acc += sq[i] * W_off[i * 128 + tid];
  soff[tid] = acc;
  __syncthreads();

  if (tid < 2) {  // per-coordinate mean / unbiased std over 64 points
    float s = 0.f;
    for (int p = 0; p < 64; ++p) s += soff[2 * p + tid];
    float mean = s * (1.f / 64.f);
    float v = 0.f;
    for (int p = 0; p < 64; ++p) {
      float d = soff[2 * p + tid] - mean;
      v += d * d;
    }
    float sd = sqrtf(v * (1.f / 63.f)) + 1e-7f;
    sstat[tid * 2]     = mean;
    sstat[tid * 2 + 1] = 1.f / (3.f * sd);
  }
  __syncthreads();

  {
    const int c = tid & 1;
    float lo  = token_roi[tok * 4 + c];
    float hi  = token_roi[tok * 4 + 2 + c];
    float ctr = 0.5f * (lo + hi);
    float wh  = hi - lo;
    spts[tid] = ctr + (soff[tid] - sstat[c * 2]) * sstat[c * 2 + 1] * wh;
  }
  __syncthreads();

  for (int it = 0; it < 32; ++it) {
    const int idx = it * 128 + tid;  // = s*64 + ch
    const int s = idx >> 6, ch = idx & 63;
    float x = spts[2 * s]     * 56.f - 0.5f;
    float y = spts[2 * s + 1] * 56.f - 0.5f;
    float x0f = floorf(x), y0f = floorf(y);
    int   x0 = (int)x0f, y0 = (int)y0f;
    float wx1 = x - x0f, wx0 = 1.f - wx1;
    float wy1 = y - y0f, wy0 = 1.f - wy1;
    const int head = s >> 4;
    const float* plane = img_feat + (((size_t)b * 256 + head * 64 + ch) * (56 * 56));
    float v = 0.f;  // zeros padding
    if ((unsigned)x0       < 56u && (unsigned)y0       < 56u) v += wx0 * wy0 * plane[y0 * 56 + x0];
    if ((unsigned)(x0 + 1) < 56u && (unsigned)y0       < 56u) v += wx1 * wy0 * plane[y0 * 56 + x0 + 1];
    if ((unsigned)x0       < 56u && (unsigned)(y0 + 1) < 56u) v += wx0 * wy1 * plane[(y0 + 1) * 56 + x0];
    if ((unsigned)(x0 + 1) < 56u && (unsigned)(y0 + 1) < 56u) v += wx1 * wy1 * plane[(y0 + 1) * 56 + x0 + 1];
    sampled_h[(size_t)tok * 4096 + idx] = (_Float16)v;
  }
}

// ---------------------------------------------------------------------------
// K2: hidden = LN(query) @ W1 + b1  (f16 out).  1 block (64 thr) per token.
// ---------------------------------------------------------------------------
__global__ void __launch_bounds__(64)
k_hidden(const float* __restrict__ query, const float* __restrict__ ln_w,
         const float* __restrict__ ln_b, const float* __restrict__ W1,
         const float* __restrict__ b1, _Float16* __restrict__ hidden_h) {
  __shared__ float sq[256];
  __shared__ float red[64], red2[64];
  const int tid = threadIdx.x;
  const int tok = blockIdx.x;
  float s = 0.f, s2 = 0.f;
  for (int i = tid; i < 256; i += 64) {
    float q = query[tok * 256 + i];
    sq[i] = q; s += q; s2 += q * q;
  }
  red[tid] = s; red2[tid] = s2;
  __syncthreads();
  for (int off = 32; off > 0; off >>= 1) {
    if (tid < off) { red[tid] += red[tid + off]; red2[tid] += red2[tid + off]; }
    __syncthreads();
  }
  const float mu   = red[0] * (1.f / 256.f);
  const float var  = red2[0] * (1.f / 256.f) - mu * mu;  // ddof=0
  const float rstd = rsqrtf(var + 1e-6f);
  for (int i = tid; i < 256; i += 64)
    sq[i] = (sq[i] - mu) * rstd * ln_w[i] + ln_b[i];
  __syncthreads();
  float acc = b1[tid];
  for (int i = 0; i < 256; ++i) acc += sq[i] * W1[i * 64 + tid];
  hidden_h[tok * 64 + tid] = (_Float16)acc;
}

// ---------------------------------------------------------------------------
// K3: fused param-gen + adaptive mixing. 8 tokens/block, 8 waves, ~129KB LDS.
//   phase1: cm = hidden @ W2p[:, :4096]  -> sP  (B-fragment layout per token)
//   phase2: x1 = gelu(sampled @ cm + mb) -> sX1 (B-fragment layout per token)
//   phase3: sm = hidden @ W2p[:, 4096:]  -> sP  (row-major per token, A use)
//   phase4: x2 = gelu(sm @ x1 + sb)      -> global f16 row-major
// ---------------------------------------------------------------------------
__global__ void __launch_bounds__(256)
k_mix(const _Float16* __restrict__ hidden_h, const _Float16* __restrict__ sampled_h,
      const _Float16* __restrict__ W2p, const float* __restrict__ b2,
      const float* __restrict__ m_beta, const float* __restrict__ s_beta,
      _Float16* __restrict__ x2_h) {
  __shared__ alignas(32) _Float16 sP [8 * 64 * 64];  // 64 KB
  __shared__ alignas(32) _Float16 sX1[8 * 64 * 64];  // 64 KB
  __shared__ alignas(32) _Float16 sH [8 * 64];       // 1 KB
  const int tid  = threadIdx.x;
  const int lane = tid & 31;
  const int wid  = tid >> 5;
  const int tok0 = blockIdx.x * 8;

  // Stage hidden slab (8x64 f16 = 1KB) to LDS via CDNA5 async copy.
  if (tid < 32) {
    const _Float16* hsrc = hidden_h + (size_t)tok0 * 64;
    async_copy_b128(hsrc + tid * 8, sH + tid * 8);
    async_copy_b128(hsrc + 256 + tid * 8, sH + 256 + tid * 8);
    wait_async0();
  }
  __syncthreads();

  // hidden A fragments (rows 0..7 = tokens, rows 8..15 zero), K = 64
  const v16h hA0 = load_a_f16(sH, 64, 0, 0,  lane, 8);
  const v16h hA1 = load_a_f16(sH, 64, 0, 32, lane, 8);

  // ---- phase 1: channel-mixing params -> fragment-major LDS ----
  for (int tile = wid; tile < 256; tile += 8) {
    const int j0 = tile * 16;
    v8f d = {};
    d = wmma_f16(hA0, load_b_packed(W2p, 0 * 512 + tile, lane), d);
    d = wmma_f16(hA1, load_b_packed(W2p, 1 * 512 + tile, lane), d);
    const float bb = b2[j0 + (lane & 15)];
    if (lane < 16) {  // rows 0..7 (tokens) live in lanes 0..15
      const int c    = j0 >> 6;     // cm row (K for phase 2)
      const int d0   = j0 & 63;     // cm col base (N for phase 2)
      const int kq   = c >> 5;
      const int lnP  = lane + (((c >> 4) & 1) << 4);
      const int base = (kq * 4 + (d0 >> 4)) * 512 + lnP * 16 + (c & 15);
#pragma unroll
      for (int r = 0; r < 8; ++r)
        sP[r * 4096 + base] = (_Float16)(d[r] + bb);
    }
  }
  __syncthreads();

  // ---- phase 2: x1 = gelu(sampled @ cm + m_beta) -> fragment-major LDS ----
  for (int tt = wid; tt < 128; tt += 8) {
    const int t = tt >> 4, pt = (tt >> 2) & 3, ct = tt & 3;
    const _Float16* Ap = sampled_h + (size_t)(tok0 + t) * 4096;
    const _Float16* Bp = &sP[t * 4096];
    v8f d = {};
    d = wmma_f16(load_a_f16(Ap, 64, pt * 16, 0,  lane, 16),
                 load_b_packed(Bp, 0 * 4 + ct, lane), d);
    d = wmma_f16(load_a_f16(Ap, 64, pt * 16, 32, lane, 16),
                 load_b_packed(Bp, 1 * 4 + ct, lane), d);
    const int   n    = ct * 16 + (lane & 15);
    const float mb   = m_beta[n];
    const int   mrow = (lane >> 4) * 8;
    v8h pk;
#pragma unroll
    for (int r = 0; r < 8; ++r) pk[r] = (_Float16)gelu_exact(d[r] + mb);
    // x1 as phase-4 B fragments: K index = p (contiguous in r) -> one b128.
    const int kq  = pt >> 1;
    const int lnP = (lane & 15) + ((pt & 1) << 4);
    *(v8h*)(&sX1[t * 4096 + (kq * 4 + ct) * 512 + lnP * 16 + mrow]) = pk;
  }
  __syncthreads();

  // ---- phase 3: spatial-mixing params -> row-major LDS (A operand) ----
  for (int tile = wid; tile < 256; tile += 8) {
    const int j0 = tile * 16;
    v8f d = {};
    d = wmma_f16(hA0, load_b_packed(W2p, 0 * 512 + 256 + tile, lane), d);
    d = wmma_f16(hA1, load_b_packed(W2p, 1 * 512 + 256 + tile, lane), d);
    const float bb = b2[4096 + j0 + (lane & 15)];
    if (lane < 16) {
#pragma unroll
      for (int r = 0; r < 8; ++r)
        sP[r * 4096 + j0 + lane] = (_Float16)(d[r] + bb);  // [t][o][p]
    }
  }
  __syncthreads();

  // ---- phase 4: x2 = gelu(sm @ x1 + s_beta[o]) -> global row-major ----
  for (int tt = wid; tt < 128; tt += 8) {
    const int t = tt >> 4, ot = (tt >> 2) & 3, ct = tt & 3;
    v8f d = {};
    d = wmma_f16(load_a_f16(&sP[t * 4096], 64, ot * 16, 0,  lane, 16),
                 load_b_packed(&sX1[t * 4096], 0 * 4 + ct, lane), d);
    d = wmma_f16(load_a_f16(&sP[t * 4096], 64, ot * 16, 32, lane, 16),
                 load_b_packed(&sX1[t * 4096], 1 * 4 + ct, lane), d);
    const int n    = ct * 16 + (lane & 15);
    const int mrow = (lane >> 4) * 8;
    _Float16* outp = x2_h + (size_t)(tok0 + t) * 4096;
#pragma unroll
    for (int r = 0; r < 8; ++r) {
      const int o = ot * 16 + mrow + r;  // s_beta broadcast per row (o)
      float v = gelu_exact(d[r] + s_beta[o]);
      outp[o * 64 + n] = (_Float16)v;
    }
  }
}

// ---------------------------------------------------------------------------
// K4: out = x2 [4096,4096] @ W_out [4096,256] + b_out.
// 128 blocks x 8 waves; wave = 16-row M tile x 4 N tiles, A reused across N.
// ---------------------------------------------------------------------------
__global__ void __launch_bounds__(256)
k_out(const _Float16* __restrict__ x2_h, const _Float16* __restrict__ Woutp,
      const float* __restrict__ b_out, float* __restrict__ out) {
  const int lane   = threadIdx.x & 31;
  const int wid    = threadIdx.x >> 5;
  const int row0   = blockIdx.x * 32 + (wid >> 2) * 16;
  const int ntBase = (wid & 3) * 4;
  v8f acc[4];
  v8f z = {};
#pragma unroll
  for (int i = 0; i < 4; ++i) acc[i] = z;

  for (int k0 = 0; k0 < 4096; k0 += 32) {
    if (k0 + 32 < 4096)  // stream A ahead (global_prefetch_b8)
      __builtin_prefetch(x2_h + (size_t)(row0 + (lane & 15)) * 4096 + k0 + 32, 0, 0);
    v16h a = load_a_f16(x2_h, 4096, row0, k0, lane, 16);
    const int kt = k0 >> 5;
#pragma unroll
    for (int i = 0; i < 4; ++i)
      acc[i] = wmma_f16(a, load_b_packed(Woutp, kt * 16 + ntBase + i, lane), acc[i]);
  }
  const int mrow = (lane >> 4) * 8;
#pragma unroll
  for (int i = 0; i < 4; ++i) {
    const int n = (ntBase + i) * 16 + (lane & 15);
    const float bb = b_out[n];
#pragma unroll
    for (int r = 0; r < 8; ++r)
      out[(size_t)(row0 + mrow + r) * 256 + n] = acc[i][r] + bb;
  }
}

// ---------------------------------------------------------------------------
extern "C" void kernel_launch(void* const* d_in, const int* in_sizes, int n_in,
                              void* d_out, int out_size, void* d_ws, size_t ws_size,
                              hipStream_t stream) {
  (void)in_sizes; (void)n_in; (void)out_size; (void)ws_size;
  const float* img_feat  = (const float*)d_in[0];
  const float* token_roi = (const float*)d_in[1];
  const float* query     = (const float*)d_in[2];
  const float* W_off     = (const float*)d_in[3];
  const float* b_off     = (const float*)d_in[4];
  const float* ln_w      = (const float*)d_in[5];
  const float* ln_b      = (const float*)d_in[6];
  const float* W1        = (const float*)d_in[7];
  const float* b1        = (const float*)d_in[8];
  const float* W2        = (const float*)d_in[9];
  const float* b2        = (const float*)d_in[10];
  const float* m_beta    = (const float*)d_in[11];
  const float* s_beta    = (const float*)d_in[12];
  const float* W_out     = (const float*)d_in[13];
  const float* b_out     = (const float*)d_in[14];
  float* out = (float*)d_out;

  _Float16* sampled_h = (_Float16*)d_ws;                        // 32 MB
  _Float16* hidden_h  = sampled_h + (size_t)4096 * 4096;        // 512 KB
  _Float16* x2_h      = hidden_h  + (size_t)4096 * 64;          // 32 MB
  _Float16* W2p       = x2_h      + (size_t)4096 * 4096;        // 1 MB
  _Float16* Woutp     = W2p       + (size_t)64 * 8192;          // 2 MB

  k_pack_b<<<128, 256, 0, stream>>>(W2, W2p, 64, 8192);
  k_pack_b<<<256, 256, 0, stream>>>(W_out, Woutp, 4096, 256);
  k_sample<<<4096, 128, 0, stream>>>(img_feat, token_roi, query, W_off, b_off,
                                     sampled_h);
  k_hidden<<<4096, 64, 0, stream>>>(query, ln_w, ln_b, W1, b1, hidden_h);
  k_mix<<<512, 256, 0, stream>>>(hidden_h, sampled_h, W2p, b2, m_beta, s_beta,
                                 x2_h);
  k_out<<<128, 256, 0, stream>>>(x2_h, Woutp, b_out, out);
}
